// Attention_65377992180588
// MI455X (gfx1250) — matrix operand order repaired and checked by reading, not compiled, not run on hardware
//
#include <hip/hip_runtime.h>
#include <stdint.h>

typedef __bf16 bf16_t;
typedef __attribute__((ext_vector_type(16))) __bf16 v16bf;
typedef __attribute__((ext_vector_type(8)))  __bf16 v8bf;
typedef __attribute__((ext_vector_type(8)))  float  v8f;
typedef __attribute__((ext_vector_type(16))) float  v16f;
typedef __attribute__((ext_vector_type(4)))  uint32_t v4u;
typedef __attribute__((ext_vector_type(8)))  uint32_t v8u;

#define HIDDEN 128
#define OUTD   128
#define BATCH  4
#define SEQ    4096
#define ROWS   (BATCH * SEQ)   // 16384
#define KT     32              // keys per flash-attention iteration

// ---------------------------------------------------------------------------
// Kernel 1: QKV projection.  out[n,d] = sum_h x[n,h] * W[d,h]  (W row-major
// [out,in], so B-fragment B[k][n] = W[n][k] reads rows of W contiguously).
// One 16x16 tile per wave, 4 x v_wmma_f32_16x16x32_bf16 along K=128.
// ---------------------------------------------------------------------------
__global__ __launch_bounds__(256) void qkv_proj_kernel(
    const float* __restrict__ x,
    const float* __restrict__ Wq,
    const float* __restrict__ Wk,
    const float* __restrict__ Wv,
    bf16_t* __restrict__ qkv /* [3][ROWS][HIDDEN] bf16 */) {

    const int lane = threadIdx.x & 31;
    const int wave = threadIdx.x >> 5;
    const int task = blockIdx.x * 8 + wave;           // 24576 tasks total

    const int tiles_per_mat = (ROWS / 16) * (HIDDEN / 16);  // 8192
    const int mat = task / tiles_per_mat;
    const int rem = task % tiles_per_mat;
    const int tm  = rem / (HIDDEN / 16);
    const int tn  = rem % (HIDDEN / 16);

    const float* W = (mat == 0) ? Wq : (mat == 1) ? Wk : Wv;
    bf16_t* dst = qkv + (size_t)mat * ROWS * HIDDEN;

    const int half = lane >> 4;
    const int l16  = lane & 15;

    const float* xrow = x + (size_t)(tm * 16 + l16) * HIDDEN;  // A: M = lane%16
    const float* wrow = W + (size_t)(tn * 16 + l16) * HIDDEN;  // B: N = lane%16

    v8f c = {};
#pragma unroll
    for (int ks = 0; ks < 4; ++ks) {
        // A 16x32 bf16, contiguous k order used here (same as B): elem i -> K = half*16 + i
        v8f fa0 = *(const v8f*)(xrow + ks * 32 + half * 16);       // 32B aligned
        v8f fa1 = *(const v8f*)(xrow + ks * 32 + half * 16 + 8);   // 32B aligned
        // B 32x16 bf16, contiguous k order used here (same as A): elem i -> K = half*16 + i
        v16f fb = *(const v16f*)(wrow + ks * 32 + half * 16);      // 64B aligned

        v16bf a, b;
#pragma unroll
        for (int i = 0; i < 8; ++i) {
            a[i]     = (bf16_t)fa0[i];
            a[8 + i] = (bf16_t)fa1[i];
        }
#pragma unroll
        for (int i = 0; i < 16; ++i)
            b[i] = (bf16_t)fb[i];

        c = __builtin_amdgcn_wmma_f32_16x16x32_bf16(false, a, false, b,
                                                    (short)0, c, false, false);
    }

    // C/D layout: VGPR r holds row M = r + 8*half, col N = lane%16
#pragma unroll
    for (int r = 0; r < 8; ++r) {
        int row = tm * 16 + r + 8 * half;
        dst[(size_t)row * HIDDEN + tn * 16 + l16] = (bf16_t)c[r];
    }
}

// ---------------------------------------------------------------------------
// Kernel 2: fused flash attention.  128 blocks = 4 batches x 32 query blocks
// of 128 rows; 8 waves/block, each wave owns 16 query rows and a full
// 16x128 fp32 accumulator (8 C tiles).  Per 32-key iteration:
//   K tile  : per-lane GLOBAL_LOAD_ASYNC_TO_LDS_B128  (ASYNCcnt)
//   V tile  : TENSOR_LOAD_TO_LDS (TDM, 2-D D#)        (TENSORcnt)
//   then S = Q K^T (8 WMMAs) -> online softmax -> P V (8 WMMAs).
// ---------------------------------------------------------------------------
__global__ __launch_bounds__(256) void attn_kernel(
    const bf16_t* __restrict__ qkv, float* __restrict__ out) {

    __shared__ __attribute__((aligned(32))) bf16_t sK [KT][HIDDEN];  // 8 KB
    __shared__ __attribute__((aligned(32))) bf16_t sV [KT][HIDDEN];  // 8 KB (TDM dest, row-major)
    __shared__ __attribute__((aligned(32))) bf16_t sVt[OUTD][KT];    // 8 KB (transposed copy)
    __shared__ __attribute__((aligned(32))) bf16_t sP [8][16][KT];   // 8 KB

    const bf16_t* qw = qkv;
    const bf16_t* kw = qkv + (size_t)ROWS * HIDDEN;
    const bf16_t* vw = qkv + (size_t)2 * ROWS * HIDDEN;

    const int lane = threadIdx.x & 31;
    const int wave = threadIdx.x >> 5;
    const int half = lane >> 4;
    const int l16  = lane & 15;

    const int b     = blockIdx.x >> 5;            // batch
    const int qblk  = blockIdx.x & 31;            // 128-row query block
    const int qrow0 = qblk * 128 + wave * 16;     // wave's first query row

    // Staging assignment: 256 threads x 32B covers a 32x128 bf16 tile.
    const int skey = threadIdx.x >> 3;            // key row 0..31
    const int sd0  = (threadIdx.x & 7) * 16;      // dim offset 0..112
    const uint32_t ldsK = (uint32_t)(uintptr_t)&sK[skey][sd0];  // LDS byte offset
    const uint32_t ldsV = (uint32_t)(uintptr_t)&sV[0][0];

    // Q fragments (A layout), K-dim 128 in 4 steps of 32 — kept in registers.
    const size_t qbase = ((size_t)b * SEQ + qrow0 + l16) * HIDDEN;
    v16bf qa[4];
#pragma unroll
    for (int ks = 0; ks < 4; ++ks) {
        v8bf q0 = *(const v8bf*)(qw + qbase + ks * 32 + half * 16);
        v8bf q1 = *(const v8bf*)(qw + qbase + ks * 32 + half * 16 + 8);
        qa[ks] = __builtin_shufflevector(q0, q1, 0, 1, 2, 3, 4, 5, 6, 7,
                                                 8, 9, 10, 11, 12, 13, 14, 15);
    }

    float m[8], s[8];
    v8f acc[8];
#pragma unroll
    for (int r = 0; r < 8; ++r) { m[r] = -1e30f; s[r] = 0.0f; acc[r] = (v8f){}; }

    const float scale = 0.088388347648318447f;   // 1/sqrt(128)

    for (int kt = 0; kt < SEQ / KT; ++kt) {
        __syncthreads();  // previous iteration's LDS reads are done

        const size_t grow = ((size_t)b * SEQ + kt * KT + skey) * HIDDEN + sd0;

        // K tile: per-lane async DMA global -> LDS (2 x 16B, ASYNCcnt-tracked).
        {
            const bf16_t* ksrc = kw + grow;
            asm volatile("global_load_async_to_lds_b128 %0, %1, off"
                         :: "v"(ldsK), "v"(ksrc) : "memory");
            asm volatile("global_load_async_to_lds_b128 %0, %1, off offset:16"
                         :: "v"(ldsK), "v"(ksrc) : "memory");
        }
        // V tile: Tensor Data Mover, one 2-D descriptor issued by wave 0.
        // D# group0: count=1 | lds_addr | global_addr[56:0] | type=2
        // D# group1: mask=0, data_size=1(2B), tensor_dim0=128, tensor_dim1=32,
        //            tile_dim0=128, tile_dim1=32, tensor_dim0_stride=128.
        if (wave == 0) {
            uint64_t gaddr = (uint64_t)(uintptr_t)(vw + ((size_t)b * SEQ + kt * KT) * HIDDEN);
            v4u g0;
            g0[0] = 1u;                                     // count=1
            g0[1] = ldsV;                                   // lds_addr
            g0[2] = (uint32_t)gaddr;                        // global_addr[31:0]
            g0[3] = (uint32_t)(gaddr >> 32) | (2u << 30);   // global_addr[56:32] | type=2
            v8u g1;
            g1[0] = 0x00010000u;                            // data_size = 1 (2 bytes)
            g1[1] = (uint32_t)HIDDEN << 16;                 // tensor_dim0 = 128
            g1[2] = (uint32_t)KT << 16;                     // tensor_dim1 = 32
            g1[3] = (uint32_t)HIDDEN << 16;                 // tile_dim0 = 128
            g1[4] = (uint32_t)KT;                           // tile_dim1 = 32 (tile_dim2 = 0)
            g1[5] = (uint32_t)HIDDEN;                       // tensor_dim0_stride = 128
            g1[6] = 0u;                                     // stride hi / dim1_stride lo
            g1[7] = 0u;
            asm volatile("tensor_load_to_lds %0, %1"
                         :: "s"(g0), "s"(g1) : "memory");
            __builtin_amdgcn_s_wait_tensorcnt(0);
        }
        // Prefetch next iteration's K/V rows into L2.
        if (kt + 1 < SEQ / KT) {
            __builtin_prefetch(kw + grow + (size_t)KT * HIDDEN, 0, 0);
            __builtin_prefetch(vw + grow + (size_t)KT * HIDDEN, 0, 0);
        }
        asm volatile("s_wait_asynccnt 0" ::: "memory");
        __syncthreads();   // sK (async) and sV (TDM) are now valid

        // Transpose V tile: LDS row-major -> LDS key-major (read 32B, scatter b16).
        {
            v16bf vv = *(const v16bf*)&sV[skey][sd0];
#pragma unroll
            for (int i = 0; i < 16; ++i)
                sVt[sd0 + i][skey] = vv[i];
        }

        // S = Q K^T : two 16x16 score tiles covering 32 keys.
        v8f sc[2];
#pragma unroll
        for (int st = 0; st < 2; ++st) {
            v8f cc = {};
#pragma unroll
            for (int ks = 0; ks < 4; ++ks) {
                // B: N = key (lane%16), elem i -> K = half*16 + i  (32B aligned)
                v16bf kb = *(const v16bf*)&sK[st * 16 + l16][ks * 32 + half * 16];
                cc = __builtin_amdgcn_wmma_f32_16x16x32_bf16(
                         false, qa[ks], false, kb, (short)0, cc, false, false);
            }
            sc[st] = cc;
        }

        // Online softmax.  Row M = r + 8*half lives across the 16 lanes of a
        // half-group; xor masks 1..8 never cross the group boundary.
        float al[8];
#pragma unroll
        for (int r = 0; r < 8; ++r) {
            float v0 = sc[0][r] * scale;
            float v1 = sc[1][r] * scale;
            float rmax = fmaxf(v0, v1);
#pragma unroll
            for (int msk = 8; msk >= 1; msk >>= 1)
                rmax = fmaxf(rmax, __shfl_xor(rmax, msk, 32));
            float newm  = fmaxf(m[r], rmax);
            float alpha = __expf(m[r] - newm);
            float p0 = __expf(v0 - newm);
            float p1 = __expf(v1 - newm);
            float rsum = p0 + p1;
#pragma unroll
            for (int msk = 8; msk >= 1; msk >>= 1)
                rsum += __shfl_xor(rsum, msk, 32);
            s[r] = s[r] * alpha + rsum;
            m[r] = newm;
            al[r] = alpha;
            // D-layout -> LDS so we can re-read P in A-layout.
            sP[wave][r + 8 * half][l16]      = (bf16_t)p0;
            sP[wave][r + 8 * half][16 + l16] = (bf16_t)p1;
        }
#pragma unroll
        for (int nt = 0; nt < 8; ++nt)
#pragma unroll
            for (int r = 0; r < 8; ++r)
                acc[nt][r] *= al[r];

        __syncthreads();  // sP and sVt visible to the P*V stage

        // P (16x32, A-layout) x V (32x128) accumulated into 8 output tiles.
        v8bf plo = *(const v8bf*)&sP[wave][l16][half * 16];
        v8bf phi = *(const v8bf*)&sP[wave][l16][half * 16 + 8];
        v16bf pa = __builtin_shufflevector(plo, phi, 0, 1, 2, 3, 4, 5, 6, 7,
                                                     8, 9, 10, 11, 12, 13, 14, 15);
#pragma unroll
        for (int nt = 0; nt < 8; ++nt) {
            // B: N = out dim (lane%16), elem i -> key = half*16 + i  (32B aligned)
            v16bf vb = *(const v16bf*)&sVt[nt * 16 + l16][half * 16];
            acc[nt] = __builtin_amdgcn_wmma_f32_16x16x32_bf16(
                          false, pa, false, vb, (short)0, acc[nt], false, false);
        }
    }

    // Epilogue: normalize by running sum and store fp32.
#pragma unroll
    for (int r = 0; r < 8; ++r) {
        float inv = 1.0f / s[r];
        int row = qrow0 + r + 8 * half;
        float* orow = out + ((size_t)b * SEQ + row) * OUTD;
#pragma unroll
        for (int nt = 0; nt < 8; ++nt)
            orow[nt * 16 + l16] = acc[nt][r] * inv;
    }
}

// ---------------------------------------------------------------------------
extern "C" void kernel_launch(void* const* d_in, const int* in_sizes, int n_in,
                              void* d_out, int out_size, void* d_ws, size_t ws_size,
                              hipStream_t stream) {
    const float* x  = (const float*)d_in[0];
    const float* Wq = (const float*)d_in[1];
    const float* Wk = (const float*)d_in[2];
    const float* Wv = (const float*)d_in[3];
    bf16_t* qkv = (bf16_t*)d_ws;     // 3 * 16384 * 128 bf16 = 12 MB
    float*  out = (float*)d_out;

    // 24576 tile-tasks / 8 waves per block
    qkv_proj_kernel<<<3072, 256, 0, stream>>>(x, Wq, Wk, Wv, qkv);
    // 4 batches x 32 query blocks
    attn_kernel<<<128, 256, 0, stream>>>(qkv, out);
}
